// ObliviousLayer_12833362280720
// MI455X (gfx1250) — matrix operand order repaired and checked by reading, not compiled, not run on hardware
//
#include <hip/hip_runtime.h>
#include <hip/hip_bf16.h>
#include <math.h>

// ---------------------------------------------------------------------------
// Problem constants (match reference)
// ---------------------------------------------------------------------------
#define DEPTH        10
#define NFEAT        512          // features == output N
#define NLEAF        1024         // leaves   == GEMM K
#define BATCH        32768
#define INV_TEMP     10.0f        // 1 / 0.1

// Workspace layout (bytes)
#define WS_PROBS_OFF 0
#define WS_PROBS_SZ  (DEPTH * NFEAT * 4)                 // 20480
#define WS_DEC_OFF   (WS_PROBS_OFF + WS_PROBS_SZ)
#define WS_DEC_SZ    (BATCH * DEPTH * 4)                 // 1310720
#define WS_BT_OFF    (WS_DEC_OFF + WS_DEC_SZ)            // 1331200 (32B aligned)

typedef __attribute__((ext_vector_type(16))) __bf16 v16bf;
typedef __attribute__((ext_vector_type(8)))  float  v8f;

// ---------------------------------------------------------------------------
// Kernel 1: probs = softmax(feature_weights, axis=1).  10 rows of 512.
// ---------------------------------------------------------------------------
__global__ __launch_bounds__(256)
void softmax_kernel(const float* __restrict__ fw, float* __restrict__ probs) {
    __shared__ float red[256];
    const int r   = blockIdx.x;        // 0..9
    const int tid = threadIdx.x;

    float m = -INFINITY;
    for (int f = tid; f < NFEAT; f += 256) m = fmaxf(m, fw[r * NFEAT + f]);
    red[tid] = m; __syncthreads();
    for (int off = 128; off > 0; off >>= 1) {
        if (tid < off) red[tid] = fmaxf(red[tid], red[tid + off]);
        __syncthreads();
    }
    const float mx = red[0]; __syncthreads();

    float s = 0.0f;
    for (int f = tid; f < NFEAT; f += 256) s += __expf(fw[r * NFEAT + f] - mx);
    red[tid] = s; __syncthreads();
    for (int off = 128; off > 0; off >>= 1) {
        if (tid < off) red[tid] += red[tid + off];
        __syncthreads();
    }
    const float inv = 1.0f / red[0];
    for (int f = tid; f < NFEAT; f += 256)
        probs[r * NFEAT + f] = __expf(fw[r * NFEAT + f] - mx) * inv;
}

// ---------------------------------------------------------------------------
// Kernel 2: pack B = leaf_values^T (K=1024 x N=512) into bf16 WMMA fragment
// tiles.  Fragment (s, nt): k in [32s,32s+32), n in [16nt,16nt+16).
// Contiguous K order (same as A): lanes 0-15 hold col n=lane, K 0..15; lanes 16-31 hold
// col n=lane-16, K 16..31.  Each lane's 16 bf16 are stored contiguously
// (32 B) so the GEMM loads one lane-fragment with two b128 loads.
// Grid: 32768 threads == 1024 fragments * 32 lanes.
// ---------------------------------------------------------------------------
__global__ __launch_bounds__(256)
void pack_b_kernel(const float* __restrict__ lv, unsigned short* __restrict__ bt) {
    const int g    = blockIdx.x * 256 + threadIdx.x;
    const int lane = g & 31;
    const int frag = g >> 5;                 // 0..1023  (= s*32 + nt)
    const int s    = frag >> 5;              // k step
    const int nt   = frag & 31;              // n tile
    const int n    = nt * 16 + (lane & 15);
    const int kb   = s * 32 + ((lane & 16) ? 16 : 0);

    const float*    src = lv + (size_t)n * NLEAF + kb;   // lv[n][k], k contiguous
    unsigned short* dst = bt + (size_t)frag * 512 + lane * 16;
#pragma unroll
    for (int j = 0; j < 16; ++j) {
        __bf16 b = (__bf16)src[j];
        dst[j] = __builtin_bit_cast(unsigned short, b);
    }
}

// ---------------------------------------------------------------------------
// Kernel 3: decisions = sigmoid((x @ probs.T - th) / T).  Wave per row.
// This is the only pass that reads x (64 MB) -> do it exactly once.
// ---------------------------------------------------------------------------
__global__ __launch_bounds__(256)
void sel_kernel(const float* __restrict__ x, const float* __restrict__ probs,
                const float* __restrict__ th, float* __restrict__ dec) {
    __shared__ float sp[DEPTH * NFEAT];                 // 20 KB
    for (int i = threadIdx.x; i < DEPTH * NFEAT; i += 256) sp[i] = probs[i];
    __syncthreads();

    const int lane = threadIdx.x & 31;
    const int w    = threadIdx.x >> 5;
    const int row  = blockIdx.x * 8 + w;
    const float* xr = x + (size_t)row * NFEAT;

    float s[DEPTH] = {};
    for (int f = lane; f < NFEAT; f += 32) {
        const float xv = xr[f];
#pragma unroll
        for (int d = 0; d < DEPTH; ++d) s[d] = fmaf(xv, sp[d * NFEAT + f], s[d]);
    }
#pragma unroll
    for (int d = 0; d < DEPTH; ++d) {
#pragma unroll
        for (int off = 16; off > 0; off >>= 1) s[d] += __shfl_xor(s[d], off, 32);
    }
    if (lane == 0) {
#pragma unroll
        for (int d = 0; d < DEPTH; ++d) {
            const float z = (s[d] - th[d]) * INV_TEMP;
            dec[(size_t)row * DEPTH + d] = 1.0f / (1.0f + __expf(-z));
        }
    }
}

// ---------------------------------------------------------------------------
// Kernel 4: out = leaf_probs @ leaf_values^T via v_wmma_f32_16x16x32_bf16.
// leaf_probs factorizes: lp[m][k] = t_hi[m][k>>5] * t_lo[m][k&31] with two
// per-row 32-entry tables built from the 10 decisions -> A fragments are
// synthesized in registers (one v_mul + cvt per element), never stored.
// Block = 8 waves; block owns 16 rows x all 512 cols; wave w owns cols
// [64w, 64w+64) = 4 accumulator tiles.  K loop: 32 steps x 4 WMMAs.
// ---------------------------------------------------------------------------
__global__ __launch_bounds__(256)
void gemm_kernel(const float* __restrict__ dec,
                 const unsigned short* __restrict__ bt,
                 float* __restrict__ out) {
    const int lane = threadIdx.x & 31;
    const int w    = threadIdx.x >> 5;
    const int m0   = blockIdx.x * 16;
    const int row  = m0 + (lane & 15);
    const bool lo  = (lane < 16);

    // Per-row decisions (lanes L and L+16 duplicate row L -> matches A layout)
    float p[DEPTH];
#pragma unroll
    for (int d = 0; d < DEPTH; ++d) p[d] = dec[(size_t)row * DEPTH + d];

    // 5/5 split product tables: index bit (4-d) <-> depth d (MSB-first codes)
    float thi[32], tlo[32];
    thi[0] = 1.0f; tlo[0] = 1.0f;
#pragma unroll
    for (int d = 0; d < 5; ++d) {
#pragma unroll
        for (int i = (1 << d) - 1; i >= 0; --i) {
            const float bh = thi[i];
            thi[2 * i + 1] = bh * p[d];
            thi[2 * i]     = bh * (1.0f - p[d]);
            const float bl = tlo[i];
            tlo[2 * i + 1] = bl * p[5 + d];
            tlo[2 * i]     = bl * (1.0f - p[5 + d]);
        }
    }

    // Pre-permute t_lo into this lane's A-fragment K order:
    //   lane<16 : K = {0..15};  lane>=16 : K = {16..31}
    float tl16[16];
#pragma unroll
    for (int j = 0; j < 16; ++j) {
        const int aI = j;                           // 0..15
        const int bI = 16 + j;                      // 16..31
        tl16[j] = lo ? tlo[aI] : tlo[bI];
    }

    v8f acc[4] = {};
    const int ntBase = w * 4;                        // wave's first global n-tile

#pragma unroll
    for (int s = 0; s < 32; ++s) {                   // K step: leaves [32s,32s+32)
        const float hs = thi[s];                     // high-bits factor (constant in step)
        v16bf aF;
#pragma unroll
        for (int j = 0; j < 16; ++j) aF[j] = (__bf16)(hs * tl16[j]);

#pragma unroll
        for (int t = 0; t < 4; ++t) {
            const v16bf bF = *(const v16bf*)(bt +
                (((size_t)(s * 32 + ntBase + t) * 32 + lane) * 16));
            acc[t] = __builtin_amdgcn_wmma_f32_16x16x32_bf16(
                false, aF, false, bF, (short)0, acc[t], false, false);
        }
    }

    // C/D layout: VGPR r -> M = r (lanes 0-15) / r+8 (lanes 16-31); N = lane&15
    const int nW    = w * 64 + (lane & 15);
    const int mBias = lo ? 0 : 8;
#pragma unroll
    for (int t = 0; t < 4; ++t) {
#pragma unroll
        for (int r = 0; r < 8; ++r) {
            out[(size_t)(m0 + r + mBias) * NFEAT + nW + t * 16] = acc[t][r];
        }
    }
}

// ---------------------------------------------------------------------------
extern "C" void kernel_launch(void* const* d_in, const int* in_sizes, int n_in,
                              void* d_out, int out_size, void* d_ws, size_t ws_size,
                              hipStream_t stream) {
    const float* x  = (const float*)d_in[0];   // (32768, 512)
    const float* fw = (const float*)d_in[1];   // (10, 512)
    const float* th = (const float*)d_in[2];   // (10,)
    const float* lv = (const float*)d_in[3];   // (512, 1024)
    float* out = (float*)d_out;                // (32768, 512)

    char* wsb = (char*)d_ws;
    float*          probs = (float*)(wsb + WS_PROBS_OFF);
    float*          dec   = (float*)(wsb + WS_DEC_OFF);
    unsigned short* bt    = (unsigned short*)(wsb + WS_BT_OFF);

    softmax_kernel<<<DEPTH, 256, 0, stream>>>(fw, probs);
    pack_b_kernel<<<(NLEAF * 32) / 256, 256, 0, stream>>>(lv, bt);   // 128 blocks
    sel_kernel<<<BATCH / 8, 256, 0, stream>>>(x, probs, th, dec);    // 4096 blocks
    gemm_kernel<<<BATCH / 16, 256, 0, stream>>>(dec, bt, out);       // 2048 blocks
}
